// Phi3FastMLP_71262097375264
// MI455X (gfx1250) — compile-verified, matrix operand order checked by reading, not run on hardware
//
#include <hip/hip_runtime.h>
#include <hip/hip_bf16.h>

typedef __attribute__((ext_vector_type(16))) _Float16 v16h;
typedef __attribute__((ext_vector_type(8)))  _Float16 v8h;
typedef __attribute__((ext_vector_type(8)))  float    v8f;
typedef __attribute__((ext_vector_type(4)))  float    v4f;

#define H_DIM 3072
#define I_DIM 8192
#define GS_Q  128
#define M_TOT 4096       // B*S = 2*2048
#define LDS_STRIDE 136   // 128 halves + 8-half pad (16B) to break bank conflicts

// ---------------------------------------------------------------------------
// Kernel 0: fp32 -> f16 convert of hidden_states (x is reused across all 64
// N-blocks of GEMM1; halving its footprint keeps the K-loop streaming weights).
// ---------------------------------------------------------------------------
__global__ __launch_bounds__(256)
void cvt_f32_to_f16_kernel(const float* __restrict__ x,
                           _Float16* __restrict__ xh, int n) {
  int i = (blockIdx.x * blockDim.x + threadIdx.x) * 4;
  if (i + 3 < n) {
    const v4f v = *(const v4f*)(x + i);
    xh[i + 0] = (_Float16)v[0];
    xh[i + 1] = (_Float16)v[1];
    xh[i + 2] = (_Float16)v[2];
    xh[i + 3] = (_Float16)v[3];
  }
}

// Cooperative fill of one 64-row x 128-col (one quant group) f16 A panel into
// LDS. 1024 16-byte chunks spread over 256 threads (4 b128 loads + 4 ds_store
// each), replacing 8x-redundant per-wave global A loads.
__device__ __forceinline__
void stage_a_panel(_Float16* __restrict__ smem,
                   const _Float16* __restrict__ a_src,  // &A[m0*K + g*128]
                   int K, int tid) {
#pragma unroll
  for (int it = 0; it < 4; ++it) {
    const int c   = tid + it * 256;   // chunk id in [0,1024)
    const int row = c >> 4;           // 64 rows
    const int cc  = (c & 15) * 8;     // half offset within row (16 chunks/row)
    const v8h v = *(const v8h*)(a_src + (size_t)row * K + cc);
    *(v8h*)(&smem[row * LDS_STRIDE + cc]) = v;
  }
}

// Build the WMMA 16-bit A fragment for one 16-row M-tile from the LDS panel.
// Per the ISA layout: lane L holds row L&15, K-chunks {kb..kb+7, kb+16..kb+23}
// with kb=(L>>4)*8 — two ds_load_b128 per fragment.
__device__ __forceinline__
v16h load_a_frag_lds(const _Float16* __restrict__ smem,
                     int mt, int lane15, int kb, int ks) {
  const _Float16* sp = &smem[(mt * 16 + lane15) * LDS_STRIDE + ks * 32 + kb];
  const v8h lo = *(const v8h*)(sp);
  const v8h hi = *(const v8h*)(sp + 16);
  v16h a;
#pragma unroll
  for (int j = 0; j < 8; ++j) { a[j] = lo[j]; a[j + 8] = hi[j]; }
  return a;
}

// ---------------------------------------------------------------------------
// Kernel 1: fused gate_up int4-dequant GEMM + SiLU.
//   act[m, i] = silu(x[m]·Wg[i]) * (x[m]·Wu[i]),  Wg = gup row i, Wu = row i+I
// Wave tile: 16 N-cols (gate + matching up) x 4 M-tiles of 16, K-steps of 32.
// Block: 8 waves -> 128 N-cols; A panel shared via LDS; grid (I/128, M/64).
// ---------------------------------------------------------------------------
__global__ __launch_bounds__(256)
void gateup_wmma_kernel(const _Float16* __restrict__ xh,
                        const int* __restrict__ qw,
                        const int* __restrict__ qz,
                        const float* __restrict__ sc,
                        _Float16* __restrict__ act) {
  __shared__ _Float16 smem[64 * LDS_STRIDE];

  const int K  = H_DIM;
  const int NG = H_DIM / GS_Q;              // 24 groups along K
  const int tid    = threadIdx.x;
  const int wave   = tid >> 5;
  const int lane   = tid & 31;
  const int lane15 = lane & 15;
  const int kb     = (lane >> 4) * 8;       // K sub-offset of this half-wave

  const int n0 = blockIdx.x * 128 + wave * 16;   // gate column base
  const int m0 = blockIdx.y * 64;                // token base

  const int ng = n0 + lane15;               // gate weight row for this lane
  const int nu = ng + I_DIM;                // up weight row
  const int* __restrict__ bg_base = qw + (size_t)ng * K;
  const int* __restrict__ bu_base = qw + (size_t)nu * K;

  v8f zero = {};
  v8f accg[4] = {zero, zero, zero, zero};
  v8f accu[4] = {zero, zero, zero, zero};

  for (int g = 0; g < NG; ++g) {
    // group-constant dequant params: w = q*scale - (zp*scale)
    const float scg = sc[(size_t)ng * NG + g];
    const float zsg = scg * (float)qz[(size_t)ng * NG + g];
    const float scu = sc[(size_t)nu * NG + g];
    const float zsu = scu * (float)qz[(size_t)nu * NG + g];

    __syncthreads();                                   // panel free to overwrite
    stage_a_panel(smem, xh + (size_t)m0 * K + g * GS_Q, K, tid);
    __syncthreads();                                   // panel ready

#pragma unroll
    for (int ks = 0; ks < 4; ++ks) {
      const int k0 = g * GS_Q + ks * 32;
      const int* __restrict__ bg = bg_base + k0 + kb;
      const int* __restrict__ bu = bu_base + k0 + kb;
      __builtin_prefetch(bg + GS_Q, 0, 1);  // next group's weight line
      __builtin_prefetch(bu + GS_Q, 0, 1);

      v16h fg, fu;
#pragma unroll
      for (int j = 0; j < 8; ++j) {
        fg[j]     = (_Float16)((float)bg[j]      * scg - zsg);
        fg[j + 8] = (_Float16)((float)bg[j + 16] * scg - zsg);
        fu[j]     = (_Float16)((float)bu[j]      * scu - zsu);
        fu[j + 8] = (_Float16)((float)bu[j + 16] * scu - zsu);
      }

#pragma unroll
      for (int mt = 0; mt < 4; ++mt) {
        const v16h a = load_a_frag_lds(smem, mt, lane15, kb, ks);
        accg[mt] = __builtin_amdgcn_wmma_f32_16x16x32_f16(
            false, a, false, fg, (short)0, accg[mt], false, false);
        accu[mt] = __builtin_amdgcn_wmma_f32_16x16x32_f16(
            false, a, false, fu, (short)0, accu[mt], false, false);
      }
    }
  }

  // Epilogue: act = silu(gate) * up, stored f16 row-major [M_TOT, I_DIM]
  // (contiguous K for kernel 2's A loads).
  const int col   = n0 + lane15;
  const int rbase = (lane >> 4) * 8;
#pragma unroll
  for (int mt = 0; mt < 4; ++mt) {
#pragma unroll
    for (int r = 0; r < 8; ++r) {
      const float gv = accg[mt][r];
      const float uv = accu[mt][r];
      const float sv = gv / (1.0f + __expf(-gv));
      const int m = m0 + mt * 16 + rbase + r;
      act[(size_t)m * I_DIM + col] = (_Float16)(sv * uv);
    }
  }
}

// ---------------------------------------------------------------------------
// Kernel 2: down int4-dequant GEMM + bias.
//   out[m, h] = act[m]·Wd[h] + bias[h]; K = I_DIM = 8192, N = H_DIM = 3072.
// ---------------------------------------------------------------------------
__global__ __launch_bounds__(256)
void down_wmma_kernel(const _Float16* __restrict__ act,
                      const int* __restrict__ qw,
                      const int* __restrict__ qz,
                      const float* __restrict__ sc,
                      const float* __restrict__ bias,
                      float* __restrict__ out) {
  __shared__ _Float16 smem[64 * LDS_STRIDE];

  const int K  = I_DIM;
  const int NG = I_DIM / GS_Q;              // 64 groups along K
  const int tid    = threadIdx.x;
  const int wave   = tid >> 5;
  const int lane   = tid & 31;
  const int lane15 = lane & 15;
  const int kb     = (lane >> 4) * 8;

  const int n0 = blockIdx.x * 128 + wave * 16;   // output hidden-dim base
  const int m0 = blockIdx.y * 64;

  const int nr = n0 + lane15;               // down weight row for this lane
  const int* __restrict__ b_base = qw + (size_t)nr * K;

  v8f zero = {};
  v8f acc[4] = {zero, zero, zero, zero};

  for (int g = 0; g < NG; ++g) {
    const float scd = sc[(size_t)nr * NG + g];
    const float zsd = scd * (float)qz[(size_t)nr * NG + g];

    __syncthreads();
    stage_a_panel(smem, act + (size_t)m0 * K + g * GS_Q, K, tid);
    __syncthreads();

#pragma unroll
    for (int ks = 0; ks < 4; ++ks) {
      const int k0 = g * GS_Q + ks * 32;
      const int* __restrict__ bp = b_base + k0 + kb;
      __builtin_prefetch(bp + GS_Q, 0, 1);

      v16h fb;
#pragma unroll
      for (int j = 0; j < 8; ++j) {
        fb[j]     = (_Float16)((float)bp[j]      * scd - zsd);
        fb[j + 8] = (_Float16)((float)bp[j + 16] * scd - zsd);
      }

#pragma unroll
      for (int mt = 0; mt < 4; ++mt) {
        const v16h a = load_a_frag_lds(smem, mt, lane15, kb, ks);
        acc[mt] = __builtin_amdgcn_wmma_f32_16x16x32_f16(
            false, a, false, fb, (short)0, acc[mt], false, false);
      }
    }
  }

  const int col   = n0 + lane15;
  const int rbase = (lane >> 4) * 8;
  const float bv  = bias[col];
#pragma unroll
  for (int mt = 0; mt < 4; ++mt) {
#pragma unroll
    for (int r = 0; r < 8; ++r) {
      const int m = m0 + mt * 16 + rbase + r;
      out[(size_t)m * H_DIM + col] = acc[mt][r] + bv;
    }
  }
}

// ---------------------------------------------------------------------------
// Launch: in-order on `stream`, deterministic, no allocation.
// Workspace layout: [0, 24 MiB) x_f16; [24 MiB, +64 MiB) act_f16.
// ---------------------------------------------------------------------------
extern "C" void kernel_launch(void* const* d_in, const int* in_sizes, int n_in,
                              void* d_out, int out_size, void* d_ws, size_t ws_size,
                              hipStream_t stream) {
  const float* hidden    = (const float*)d_in[0];
  const int*   gup_qw    = (const int*)  d_in[1];
  const int*   gup_qz    = (const int*)  d_in[2];
  const float* gup_sc    = (const float*)d_in[3];
  const int*   down_qw   = (const int*)  d_in[4];
  const int*   down_qz   = (const int*)  d_in[5];
  const float* down_sc   = (const float*)d_in[6];
  const float* down_bias = (const float*)d_in[7];
  float*       out       = (float*)d_out;

  _Float16* xh  = (_Float16*)d_ws;                                      // 4096*3072 f16
  _Float16* act = (_Float16*)((char*)d_ws + (size_t)M_TOT * H_DIM * 2); // 4096*8192 f16

  // Kernel 0: convert activations to f16.
  {
    const int n = M_TOT * H_DIM;                  // 12,582,912
    const int blocks = n / (256 * 4);             // exact: 12288
    cvt_f32_to_f16_kernel<<<blocks, 256, 0, stream>>>(hidden, xh, n);
  }

  // Kernel 1: gate_up GEMM + SiLU fusion. Grid: (I/128, M/64) = (64, 64).
  {
    dim3 grid(I_DIM / 128, M_TOT / 64);
    gateup_wmma_kernel<<<grid, 256, 0, stream>>>(xh, gup_qw, gup_qz, gup_sc, act);
  }

  // Kernel 2: down GEMM + bias. Grid: (H/128, M/64) = (24, 64).
  {
    dim3 grid(H_DIM / 128, M_TOT / 64);
    down_wmma_kernel<<<grid, 256, 0, stream>>>(act, down_qw, down_qz, down_sc,
                                               down_bias, out);
  }
}
